// BlockSparseTopkLinear_82798379532752
// MI455X (gfx1250) — matrix operand order repaired and checked by reading, not compiled, not run on hardware
//
#include <hip/hip_runtime.h>

typedef __attribute__((ext_vector_type(16))) _Float16 v16h;
typedef __attribute__((ext_vector_type(8)))  _Float16 v8h;
typedef __attribute__((ext_vector_type(4)))  _Float16 v4h;
typedef __attribute__((ext_vector_type(8)))  float    v8f;
typedef __attribute__((ext_vector_type(4)))  float    f32x4;

#define HIDDEN 4096
#define BSZ    8192
#define MB     128   // BSZ / 64 row-blocks
#define KBLK   64    // HIDDEN / 64 column-blocks
#define SEL    16    // ceil(0.25 * 64)
#define LDS_STRIDE 72             // 64 halves + 8 pad (144B rows, 16B aligned)
#define ABUF (64  * LDS_STRIDE)   // halves per A buffer
#define BBUF (256 * LDS_STRIDE)   // halves per B buffer

// ---------------------------------------------------------------------------
// CDNA5 async global->LDS (ASYNCcnt-tracked), GV mode per CDNA5 ISA:
//   mem_addr = VGPR_u64 ; LDS byte address per lane in vdst VGPR.
// ---------------------------------------------------------------------------
__device__ __forceinline__ void async_ld_b128(unsigned lds_addr,
                                              unsigned long long gaddr)
{
    asm volatile("global_load_async_to_lds_b128 %0, %1, off"
                 :: "v"(lds_addr), "v"(gaddr)
                 : "memory");
}
#define WAIT_ASYNC(N) asm volatile("s_wait_asynccnt %0" :: "i"(N) : "memory")

__device__ __forceinline__ unsigned lds_lo32(const void* p)
{
    // flat LDS pointers: addr[31:0] is the wave-relative LDS byte offset
    return (unsigned)(unsigned long long)p;
}

// ---------------------------------------------------------------------------
// Kernel 1: per-row-block |x| stats + fused f32->f16 conversion + exact top-16
// ---------------------------------------------------------------------------
__global__ __launch_bounds__(256) void stats_convert_topk(
    const float* __restrict__ x, _Float16* __restrict__ xh, int* __restrict__ sel)
{
    const int mb  = blockIdx.x;
    const int tid = threadIdx.x;
    __shared__ float smag[KBLK];
    __shared__ int   srank[KBLK];
    if (tid < KBLK) smag[tid] = 0.0f;
    __syncthreads();

    const size_t base = (size_t)mb * 64 * HIDDEN;
    float acc[4] = {0.0f, 0.0f, 0.0f, 0.0f};
    for (int r = 0; r < 64; ++r) {
#pragma unroll
        for (int j = 0; j < 4; ++j) {
            const int    c4  = tid + 256 * j;
            const size_t off = base + (size_t)r * HIDDEN + (size_t)c4 * 4;
            f32x4 v = *(const f32x4*)(x + off);
            acc[j] += __builtin_fabsf(v.x) + __builtin_fabsf(v.y) +
                      __builtin_fabsf(v.z) + __builtin_fabsf(v.w);
            v4h hv = { (_Float16)v.x, (_Float16)v.y, (_Float16)v.z, (_Float16)v.w };
            *(v4h*)(xh + off) = hv;
        }
    }
#pragma unroll
    for (int j = 0; j < 4; ++j)
        atomicAdd(&smag[(tid >> 4) + 16 * j], acc[j]);   // kb = (tid + 256j)/16
    __syncthreads();

    if (tid < KBLK) {   // rank: larger mag first, ties -> lower index (jax top_k)
        const float mi = smag[tid];
        int rank = 0;
        for (int j = 0; j < KBLK; ++j) {
            const float mj = smag[j];
            rank += (mj > mi) || (mj == mi && j < tid);
        }
        srank[tid] = rank;
    }
    __syncthreads();
    if (tid < KBLK && srank[tid] < SEL) {
        int pos = 0;
        for (int j = 0; j < tid; ++j) pos += (srank[j] < SEL);
        sel[mb * SEL + pos] = tid;
    }
}

// ---------------------------------------------------------------------------
// Kernel 2: weight f32 [k][n] -> f16 transposed wT [n][k]  (64x64 LDS tiles)
// ---------------------------------------------------------------------------
__global__ __launch_bounds__(256) void wtrans(
    const float* __restrict__ w, _Float16* __restrict__ wT)
{
    const int n0  = blockIdx.x * 64;
    const int k0  = blockIdx.y * 64;
    const int tid = threadIdx.x;
    __shared__ __align__(16) _Float16 t[64 * LDS_STRIDE];

#pragma unroll
    for (int q = 0; q < 4; ++q) {
        const int ch = tid + 256 * q;
        const int r  = ch >> 4;
        const int c4 = ch & 15;
        f32x4 v = *(const f32x4*)(w + (size_t)(k0 + r) * HIDDEN + n0 + c4 * 4);
        t[(c4 * 4 + 0) * LDS_STRIDE + r] = (_Float16)v.x;
        t[(c4 * 4 + 1) * LDS_STRIDE + r] = (_Float16)v.y;
        t[(c4 * 4 + 2) * LDS_STRIDE + r] = (_Float16)v.z;
        t[(c4 * 4 + 3) * LDS_STRIDE + r] = (_Float16)v.w;
    }
    __syncthreads();
#pragma unroll
    for (int q = 0; q < 2; ++q) {
        const int ch = tid + 256 * q;
        const int r  = ch >> 3;
        const int c  = ch & 7;
        *(f32x4*)(wT + (size_t)(n0 + r) * HIDDEN + k0 + c * 8) =
            *(const f32x4*)(t + r * LDS_STRIDE + c * 8);
    }
}

// ---------------------------------------------------------------------------
// Kernel 3: block-sparse GEMM, double-buffered async global->LDS staging.
// WG tile 64(M) x 256(N), 8 wave32s of 32x64; only selected k-blocks visited.
// ---------------------------------------------------------------------------
union FragU { v16h v; v8h p[2]; };

__global__ __launch_bounds__(256) void bsgemm(
    const _Float16* __restrict__ xh, const _Float16* __restrict__ wT,
    const int* __restrict__ sel, float* __restrict__ out)
{
    const int nb   = blockIdx.x;        // 0..15
    const int mb   = blockIdx.y;        // 0..127
    const int tid  = threadIdx.x;
    const int lane = tid & 31;
    const int wid  = tid >> 5;
    const int wm   = wid & 1;           // wave row   (2 x 32 rows)
    const int wn   = wid >> 1;          // wave col   (4 x 64 cols)
    const int h    = lane >> 4;         // half-wave select
    const int l    = lane & 15;

    __shared__ __align__(16) _Float16 sA[2 * ABUF];   // x tile  [m][k], x2 buf
    __shared__ __align__(16) _Float16 sB[2 * BBUF];   // wT tile [n][k], x2 buf

    const int mbase = mb * 64;
    const int nbase = nb * 256;
    const int* sp   = sel + mb * SEL;

    // per-thread static addressing: full 64-bit global base per chunk (GV mode)
    unsigned long long gA[2], gB[8];
    unsigned ldsA[2], ldsB[8];
    const unsigned baseA = lds_lo32(sA);
    const unsigned baseB = lds_lo32(sB);
#pragma unroll
    for (int q = 0; q < 2; ++q) {
        const int ch = tid + 256 * q;
        const int r = ch >> 3, c = ch & 7;
        gA[q]   = (unsigned long long)(xh + (size_t)(mbase + r) * HIDDEN + c * 8);
        ldsA[q] = baseA + (unsigned)((r * LDS_STRIDE + c * 8) * 2);
    }
#pragma unroll
    for (int q = 0; q < 8; ++q) {
        const int ch = tid + 256 * q;
        const int r = ch >> 3, c = ch & 7;
        gB[q]   = (unsigned long long)(wT + (size_t)(nbase + r) * HIDDEN + c * 8);
        ldsB[q] = baseB + (unsigned)((r * LDS_STRIDE + c * 8) * 2);
    }

    auto issue_tile = [&](int buf, int kb) {
        const unsigned kbyte = (unsigned)(kb * 128);           // kb*64 halves
        const unsigned ao = buf ? (unsigned)(ABUF * 2) : 0u;
        const unsigned bo = buf ? (unsigned)(BBUF * 2) : 0u;
#pragma unroll
        for (int q = 0; q < 2; ++q) async_ld_b128(ldsA[q] + ao, gA[q] + kbyte);
#pragma unroll
        for (int q = 0; q < 8; ++q) async_ld_b128(ldsB[q] + bo, gB[q] + kbyte);
    };

    v8f acc[2][4];
#pragma unroll
    for (int i = 0; i < 2; ++i)
#pragma unroll
        for (int j = 0; j < 4; ++j) {
            v8f z = {0.f, 0.f, 0.f, 0.f, 0.f, 0.f, 0.f, 0.f};
            acc[i][j] = z;
        }

    issue_tile(0, sp[0]);                                 // prologue prefetch

    for (int kk = 0; kk < SEL; ++kk) {
        const int cur = kk & 1;
        if (kk + 1 < SEL) {
            issue_tile(cur ^ 1, sp[kk + 1]);              // prefetch next buffer
            WAIT_ASYNC(10);                               // oldest 10 (= cur) done
        } else {
            WAIT_ASYNC(0);
        }
        __syncthreads();                                  // all waves' tiles in LDS

        const _Float16* cA = sA + cur * ABUF;
        const _Float16* cB = sB + cur * BBUF;
#pragma unroll
        for (int ks = 0; ks < 2; ++ks) {                  // 2 x K=32
            v16h a[2], b[4];
#pragma unroll
            for (int i = 0; i < 2; ++i) {
                // A 16-bit layout: lanes 0-15 K=0..7/16..23, lanes 16-31 +8
                const int row = wm * 32 + i * 16 + l;
                const int o   = row * LDS_STRIDE + ks * 32 + h * 8;
                FragU u;
                u.p[0] = *(const v8h*)(cA + o);
                u.p[1] = *(const v8h*)(cA + o + 16);
                a[i] = u.v;
            }
#pragma unroll
            for (int j = 0; j < 4; ++j) {
                // B 16-bit layout: lanes 0-15 K=0..7/16..23, lanes 16-31 +8
                const int row = wn * 64 + j * 16 + l;
                const int o   = row * LDS_STRIDE + ks * 32 + h * 8;
                FragU u;
                u.p[0] = *(const v8h*)(cB + o);
                u.p[1] = *(const v8h*)(cB + o + 16);
                b[j] = u.v;
            }
#pragma unroll
            for (int i = 0; i < 2; ++i)
#pragma unroll
                for (int j = 0; j < 4; ++j)
                    acc[i][j] = __builtin_amdgcn_wmma_f32_16x16x32_f16(
                        false, a[i], false, b[j], (short)0, acc[i][j],
                        false, false);
        }
        __syncthreads();                                  // buffer reuse fence
    }

    // C/D layout: VGPR g holds M = g + 8*h, N = lane&15
#pragma unroll
    for (int i = 0; i < 2; ++i) {
        const int row0 = mbase + wm * 32 + i * 16 + h * 8;
#pragma unroll
        for (int j = 0; j < 4; ++j) {
            const int col = nbase + wn * 64 + j * 16 + l;
#pragma unroll
            for (int g = 0; g < 8; ++g)
                out[(size_t)(row0 + g) * HIDDEN + col] = acc[i][j][g];
        }
    }
}

// ---------------------------------------------------------------------------
extern "C" void kernel_launch(void* const* d_in, const int* in_sizes, int n_in,
                              void* d_out, int out_size, void* d_ws, size_t ws_size,
                              hipStream_t stream)
{
    const float* x = (const float*)d_in[0];       // [8192, 4096] f32
    const float* w = (const float*)d_in[1];       // [4096, 4096] f32
    float* out     = (float*)d_out;               // [8192, 4096] f32

    char* wsb      = (char*)d_ws;
    _Float16* xh   = (_Float16*)wsb;                               // 64 MiB
    _Float16* wT   = (_Float16*)(wsb + (size_t)64 * 1024 * 1024);  // 32 MiB
    int*      sel  = (int*)(wsb + (size_t)96 * 1024 * 1024);       // 8 KiB

    stats_convert_topk<<<dim3(MB), dim3(256), 0, stream>>>(x, xh, sel);
    wtrans<<<dim3(HIDDEN / 64, HIDDEN / 64), dim3(256), 0, stream>>>(w, wT);
    bsgemm<<<dim3(HIDDEN / 256, MB), dim3(256), 0, stream>>>(xh, wT, sel, out);

    (void)in_sizes; (void)n_in; (void)out_size; (void)ws_size;
}
